// ConditionalRandomField_16836271800944
// MI455X (gfx1250) — matrix operand order repaired and checked by reading, not compiled, not run on hardware
//
#include <hip/hip_runtime.h>
#include <hip/hip_bf16.h>

#define B_ 128
#define T_ 512
#define K_ 256
#define KP 264   // padded bf16 row stride for pS (avoids LDS bank conflicts on b128 A-frag loads)

typedef __attribute__((ext_vector_type(16))) __bf16 v16bf;
typedef __attribute__((ext_vector_type(8)))  float  v8f;
typedef __attribute__((ext_vector_type(4)))  unsigned int v4u;

union Frag16 { v16bf bf; v4u u[2]; };

// ---------------- mask kernel: one wave per (b,t), AND-reduce 256 elems ----------------
__global__ void k_mask(const float* __restrict__ yp, float* __restrict__ maskW) {
    int w = blockIdx.x * 8 + (threadIdx.x >> 5);   // (b*T + t), w < B*T
    int lane = threadIdx.x & 31;
    const float4* p = (const float4*)(yp + (size_t)w * K_ + lane * 8);
    float4 a = p[0], b = p[1];
    bool ok = a.x > -1000000.0f && a.y > -1000000.0f && a.z > -1000000.0f && a.w > -1000000.0f &&
              b.x > -1000000.0f && b.y > -1000000.0f && b.z > -1000000.0f && b.w > -1000000.0f;
    int allok = __all(ok);
    if (lane == 0) maskW[w] = allok ? 1.0f : 0.0f;
}

// ---------------- point + transition score per batch ----------------
__global__ void k_scores(const float* __restrict__ yp, const float* __restrict__ trans,
                         const int* __restrict__ ytrue, const float* __restrict__ maskW,
                         float* __restrict__ targetW) {
    int b = blockIdx.x;
    int tid = threadIdx.x;
    float acc = 0.0f;
    for (int t = tid; t < T_; t += 256) {
        float m = maskW[b * T_ + t];
        int yt = ytrue[b * T_ + t];
        acc += m * yp[((size_t)(b * T_ + t)) * K_ + yt];
    }
    for (int t = tid; t < T_ - 1; t += 256) {
        float m01 = maskW[b * T_ + t] * maskW[b * T_ + t + 1];
        acc += m01 * trans[ytrue[b * T_ + t] * K_ + ytrue[b * T_ + t + 1]];
    }
    __shared__ float red[256];
    red[tid] = acc; __syncthreads();
    for (int s = 128; s > 0; s >>= 1) { if (tid < s) red[tid] += red[tid + s]; __syncthreads(); }
    if (tid == 0) targetW[b] = red[0];
}

// ---------------- exp(trans) pre-laid-out in WMMA B-fragment order ----------------
// B 32x16 bf16 frag: lane l holds column n = tile*16 + (l&15), 16 contiguous K values
// starting at k0 = c*32 + (l>>4)*16.  Stored as efrag[((jt*8+c)*32 + l)*16 + e].
__global__ void k_efrag(const float* __restrict__ trans, __bf16* __restrict__ efrag) {
    int idx = blockIdx.x * 256 + threadIdx.x;   // < 65536
    int jt = idx >> 12;
    int c  = (idx >> 9) & 7;
    int l  = (idx >> 4) & 31;
    int e  = idx & 15;
    int n  = jt * 16 + (l & 15);
    int k  = c * 32 + ((l >> 4) * 16) + e;
    efrag[idx] = (__bf16)__expf(trans[k * K_ + n]);
}

// ---------------- forward scan: 16 batches/block, 16 waves, one j-tile per wave ----------------
__global__ void __launch_bounds__(512) k_forward(const float* __restrict__ yp,
                                                 const float* __restrict__ maskW,
                                                 const float* __restrict__ targetW,
                                                 const __bf16* __restrict__ efrag,
                                                 float* __restrict__ out) {
    __shared__ float  alphaS[16 * K_];     // 16 KiB
    __shared__ __bf16 pS[16 * KP];         // ~8 KiB (padded rows)
    __shared__ float  amaxS[16];

    const int tid  = threadIdx.x;
    const int lane = tid & 31;
    const int wid  = tid >> 5;            // wave id = j-tile id = reduction row
    const int bg   = blockIdx.x * 16;

    // Preload loop-invariant E B-fragments into registers (8 chunks of K => 64 VGPRs)
    Frag16 Bf[8];
    #pragma unroll
    for (int c = 0; c < 8; ++c) {
        const v4u* bp = (const v4u*)(efrag + (((size_t)(wid * 8 + c)) * 32 + lane) * 16);
        Bf[c].u[0] = bp[0];
        Bf[c].u[1] = bp[1];
    }

    // alpha init: alpha[b] = mask[b,0] * y_pred[b,0,:]
    {
        float m0 = maskW[(size_t)(bg + wid) * T_];
        const float* row = yp + (size_t)(bg + wid) * T_ * K_;
        #pragma unroll
        for (int e = 0; e < 8; ++e) {
            int j = e * 32 + lane;
            alphaS[wid * K_ + j] = m0 * row[j];
        }
    }
    __syncthreads();

    const int mrowBase = (lane >> 4) * 8;   // C/D layout: M = v + 8*(lane>=16)
    const int ncol = lane & 15;
    const int jcol = wid * 16 + ncol;       // output column owned by this thread
    const int arow = lane & 15;             // A layout: M = lane&15

    for (int t = 1; t < T_; ++t) {
        // issue x / mask loads early; consumed only after the WMMA chain
        float xv[8], mv[8];
        #pragma unroll
        for (int v = 0; v < 8; ++v) {
            int m = mrowBase + v;
            xv[v] = yp[(((size_t)(bg + m)) * T_ + t) * K_ + jcol];
            mv[v] = maskW[(size_t)(bg + m) * T_ + t];
        }

        // phase A: per-wave amax over its row + p = exp(alpha - amax) -> bf16 LDS
        float av[8];
        float mx = -3.402823466e38f;
        #pragma unroll
        for (int e = 0; e < 8; ++e) {
            av[e] = alphaS[wid * K_ + e * 32 + lane];
            mx = fmaxf(mx, av[e]);
        }
        #pragma unroll
        for (int off = 16; off > 0; off >>= 1)
            mx = fmaxf(mx, __shfl_xor(mx, off));
        if (lane == 0) amaxS[wid] = mx;
        #pragma unroll
        for (int e = 0; e < 8; ++e)
            pS[wid * KP + e * 32 + lane] = (__bf16)__expf(av[e] - mx);
        __syncthreads();

        // phase B: r = P(16x256) x E(256x16) via 8 chained bf16 WMMAs
        v8f acc = {0.f, 0.f, 0.f, 0.f, 0.f, 0.f, 0.f, 0.f};
        #pragma unroll
        for (int c = 0; c < 8; ++c) {
            int k0 = c * 32 + (lane >> 4) * 16;  // A 16x32 layout: 16 contiguous K per lane half
            Frag16 Af;
            Af.u[0] = *(const v4u*)(pS + arow * KP + k0);
            Af.u[1] = *(const v4u*)(pS + arow * KP + k0 + 8);
            acc = __builtin_amdgcn_wmma_f32_16x16x32_bf16(
                      false, Af.bf, false, Bf[c].bf, (short)0, acc, false, false);
        }

        // alpha update: new = amax + log(r) + x ; blend with mask
        #pragma unroll
        for (int v = 0; v < 8; ++v) {
            int m = mrowBase + v;
            float amax = amaxS[m];
            float old  = alphaS[m * K_ + jcol];
            float nw   = amax + __logf(acc[v]) + xv[v];
            alphaS[m * K_ + jcol] = mv[v] * nw + (1.0f - mv[v]) * old;
        }
        __syncthreads();
    }

    // final logsumexp per batch row (one wave per row) and output
    float av[8];
    float mx = -3.402823466e38f;
    #pragma unroll
    for (int e = 0; e < 8; ++e) {
        av[e] = alphaS[wid * K_ + e * 32 + lane];
        mx = fmaxf(mx, av[e]);
    }
    #pragma unroll
    for (int off = 16; off > 0; off >>= 1)
        mx = fmaxf(mx, __shfl_xor(mx, off));
    float s = 0.0f;
    #pragma unroll
    for (int e = 0; e < 8; ++e) s += __expf(av[e] - mx);
    #pragma unroll
    for (int off = 16; off > 0; off >>= 1)
        s += __shfl_xor(s, off);
    if (lane == 0) out[bg + wid] = mx + __logf(s) - targetW[bg + wid];
}

extern "C" void kernel_launch(void* const* d_in, const int* in_sizes, int n_in,
                              void* d_out, int out_size, void* d_ws, size_t ws_size,
                              hipStream_t stream) {
    const float* y_pred = (const float*)d_in[0];
    const float* trans  = (const float*)d_in[1];
    const int*   y_true = (const int*)d_in[2];
    float* out = (float*)d_out;

    // ws layout: [0,128K) E fragments (bf16), [128K,384K) mask (f32), then target (f32[B])
    __bf16* efrag  = (__bf16*)d_ws;
    float* maskW   = (float*)((char*)d_ws + 131072);
    float* targetW = (float*)((char*)d_ws + 131072 + (size_t)B_ * T_ * 4);

    k_mask  <<<dim3((B_ * T_) / 8), dim3(256), 0, stream>>>(y_pred, maskW);
    k_efrag <<<dim3(256),           dim3(256), 0, stream>>>(trans, efrag);
    k_scores<<<dim3(B_),            dim3(256), 0, stream>>>(y_pred, trans, y_true, maskW, targetW);
    k_forward<<<dim3(B_ / 16),      dim3(512), 0, stream>>>(y_pred, maskW, targetW, efrag, out);
}
